// network_model_76716705841865
// MI455X (gfx1250) — matrix operand order repaired and checked by reading, not compiled, not run on hardware
//
#include <hip/hip_runtime.h>

typedef __attribute__((ext_vector_type(16))) _Float16 v16h;
typedef __attribute__((ext_vector_type(8)))  _Float16 v8h;
typedef __attribute__((ext_vector_type(8)))  float    v8f;

#define B_    64
#define T_    128
#define S_    100
#define H_    64
#define SEGN  96
#define NSEG  (B_*SEGN)     /* 6144 rows */
#define IH_   448           /* 7*H */
#define TEMPORAL 127
#define KSEL  122
#define N1    (KSEL*B_*S_)  /* 780800 */

/* topology table layout (ints) */
#define TAB_BN      0    /* 7 (+1 pad) */
#define TAB_SEGIDX  8    /* 100 */
#define TAB_UNITOF  108  /* 100 */
#define TAB_SEGBLK  208  /* 96  */
#define TAB_INMAP   304  /* 100 */
#define TAB_AFTDST  404  /* 100 */
#define TAB_BEFDST  504  /* 100 */
#define TAB_TOTAL   604

__device__ __forceinline__ float sigm(float x) { return 1.f / (1.f + __expf(-x)); }

__device__ __forceinline__ v8f wmma32(v16h a, v16h b, v8f c) {
  return __builtin_amdgcn_wmma_f32_16x16x32_f16(false, a, false, b, (short)0, c, false, false);
}

/* A fragment (16x32 f16) from fp32 row-major [rows][stride], rows at rbase, K chunk at kb */
__device__ __forceinline__ v16h frag_a_f32(const float* A, int rbase, int stride, int kb) {
  int lane = threadIdx.x & 31;
  int m  = lane & 15;
  int ko = (lane < 16) ? 0 : 8;
  const float* row = A + (size_t)(rbase + m) * stride + kb;
  v16h a;
#pragma unroll
  for (int i = 0; i < 8; ++i) {
    a[i]     = (_Float16)row[ko + i];
    a[8 + i] = (_Float16)row[ko + 16 + i];
  }
  return a;
}

/* B fragment (32x16 f16): B[k][n] = W[nbase+n][kb+k], W row-major [N][Ktot] f16 */
__device__ __forceinline__ v16h frag_b_h(const _Float16* W, int nbase, int Ktot, int kb) {
  int lane = threadIdx.x & 31;
  int n  = lane & 15;
  int ko = (lane < 16) ? 0 : 8;
  const _Float16* p = W + (size_t)(nbase + n) * Ktot + kb + ko;
  v8h lo = *(const v8h*)p;
  v8h hi = *(const v8h*)(p + 16);
  v16h b;
#pragma unroll
  for (int i = 0; i < 8; ++i) { b[i] = lo[i]; b[8 + i] = hi[i]; }
  return b;
}

/* x_seg A fragment: K=3 valid (padded to 32) gathered via seg_block table */
__device__ __forceinline__ v16h frag_a_xseg(const float* co, int t, int rbase, const int* tab) {
  int lane = threadIdx.x & 31;
  int m  = lane & 15;
  int ko = (lane < 16) ? 0 : 8;
  int row = rbase + m;
  int b = row / SEGN, ks = row % SEGN;
  int pos = tab[TAB_SEGBLK + ks];
  const float* x = co + (((size_t)b * T_ + t) * S_ + pos) * 3;
  v16h a;
#pragma unroll
  for (int i = 0; i < 16; ++i) a[i] = (_Float16)0.f;
#pragma unroll
  for (int i = 0; i < 8; ++i) {
    int k0 = ko + i;
    if (k0 < 3) a[i] = (_Float16)x[k0];
  }
  return a;
}

/* x_int A fragment: K=21 valid (padded to 32); j = u*3+c over bn positions */
__device__ __forceinline__ v16h frag_a_xint(const float* co, int t, int rbase, const int* tab) {
  int lane = threadIdx.x & 31;
  int m  = lane & 15;
  int ko = (lane < 16) ? 0 : 8;
  int b = rbase + m;
  v16h a;
#pragma unroll
  for (int i = 0; i < 16; ++i) a[i] = (_Float16)0.f;
#pragma unroll
  for (int i = 0; i < 8; ++i) {
    int k0 = ko + i, k1 = ko + 16 + i;
    if (k0 < 21) {
      int u = k0 / 3, c = k0 % 3;
      a[i] = (_Float16)co[(((size_t)b * T_ + t) * S_ + tab[TAB_BN + u]) * 3 + c];
    }
    if (k1 < 21) {
      int u = k1 / 3, c = k1 % 3;
      a[8 + i] = (_Float16)co[(((size_t)b * T_ + t) * S_ + tab[TAB_BN + u]) * 3 + c];
    }
  }
  return a;
}

/* sel A fragment from d_out h_outputs region: row -> (k,b,s), time index k+1 */
__device__ __forceinline__ v16h frag_a_sel(const float* hout, int rbase, int kb) {
  int lane = threadIdx.x & 31;
  int m  = lane & 15;
  int ko = (lane < 16) ? 0 : 8;
  int row = rbase + m;
  int k = row / (B_ * S_);
  int rem = row % (B_ * S_);
  int b = rem / S_, s = rem % S_;
  const float* p = hout + (((size_t)b * 123 + (k + 1)) * S_ + s) * H_ + kb;
  v16h a;
#pragma unroll
  for (int i = 0; i < 8; ++i) {
    a[i]     = (_Float16)p[ko + i];
    a[8 + i] = (_Float16)p[ko + 16 + i];
  }
  return a;
}

/* ---------------- topology ---------------- */
__device__ __forceinline__ bool in_set(int v, const int* a, int n) {
  for (int i = 0; i < n; ++i) if (a[i] == v) return true;
  return false;
}

__global__ void k_topology(const long long* bn64, int* tab) {
  if (threadIdx.x != 0 || blockIdx.x != 0) return;
  int bn[7];
  for (int i = 0; i < 7; ++i) { bn[i] = (int)bn64[i]; tab[TAB_BN + i] = bn[i]; }
  tab[TAB_BN + 7] = 0;
  int interP[4] = { bn[1], bn[3], bn[4], bn[6] };
  for (int s = 0; s < S_; ++s) {
    tab[TAB_SEGIDX + s] = -1; tab[TAB_UNITOF + s] = -1;
    tab[TAB_INMAP + s] = -1;  tab[TAB_AFTDST + s] = -1; tab[TAB_BEFDST + s] = -1;
  }
  for (int u = 0; u < 7; ++u) tab[TAB_UNITOF + bn[u]] = u;
  int k = 0;
  for (int s = 0; s < S_; ++s)
    if (!in_set(s, interP, 4)) { tab[TAB_SEGIDX + s] = k; tab[TAB_SEGBLK + k] = s; k++; }
  /* h_after: seg indices 0..94 <- positions not in {0,bn1+1,bn4,bn5,bn6} */
  int notA[5] = { 0, bn[1] + 1, bn[4], bn[5], bn[6] };
  k = 0;
  for (int s = 0; s < S_; ++s)
    if (!in_set(s, notA, 5)) { tab[TAB_AFTDST + s] = k; k++; }
  /* h_before: seg indices 1..95 skipping bn1 <- positions not in {bn0,bn1,bn2,bn3,bn6-1,bn6} */
  int notB[6] = { bn[0], bn[1], bn[2], bn[3], bn[6] - 1, bn[6] };
  int kseg = 1;
  for (int s = 0; s < S_; ++s) {
    if (in_set(s, notB, 6)) continue;
    if (kseg == bn[1]) kseg++;
    if (kseg >= SEGN) break;
    tab[TAB_BEFDST + s] = kseg; kseg++;
  }
  /* In pairing: after_block[i] <- before_block[i] */
  int notAS[4] = { 0, bn[1] + 1, bn[4], bn[6] };
  int notBS[4] = { bn[1], bn[3], bn[6] - 1, bn[6] };
  int ca = 0, cb = 0;
  for (int c = 0; c < SEGN; ++c) {
    while (ca < S_ && in_set(ca, notAS, 4)) ca++;
    while (cb < S_ && in_set(cb, notBS, 4)) cb++;
    if (ca >= S_ || cb >= S_) break;
    tab[TAB_INMAP + ca] = cb; ca++; cb++;
  }
}

/* ---------------- weight prep (f32 -> f16, K-pad input weights to 32) -------- */
__global__ void k_prepw(const float* siW, const float* sfW,
                        const float* segWih, const float* segWhh,
                        const float* intWih, const float* intWhh,
                        const float* outW1,
                        _Float16* siWh, _Float16* sfWh, _Float16* segWihh,
                        _Float16* segWhhh, _Float16* intWihh, _Float16* intWhhh,
                        _Float16* outW1h) {
  int i = blockIdx.x * blockDim.x + threadIdx.x;
  int st = gridDim.x * blockDim.x;
  for (int j = i; j < 64 * 128; j += st) { siWh[j] = (_Float16)siW[j]; sfWh[j] = (_Float16)sfW[j]; }
  for (int j = i; j < 256 * 32; j += st) {
    int g = j / 32, kk = j % 32;
    segWihh[j] = (kk < 3) ? (_Float16)segWih[g * 3 + kk] : (_Float16)0.f;
  }
  for (int j = i; j < 256 * 64; j += st) segWhhh[j] = (_Float16)segWhh[j];
  for (int j = i; j < 1792 * 32; j += st) {
    int g = j / 32, kk = j % 32;
    intWihh[j] = (kk < 21) ? (_Float16)intWih[g * 21 + kk] : (_Float16)0.f;
  }
  for (int j = i; j < 1792 * 448; j += st) intWhhh[j] = (_Float16)intWhh[j];
  for (int j = i; j < 128 * 64; j += st) outW1h[j] = (_Float16)outW1[j];
}

__global__ void k_zero(float* p, long n) {
  long i = (long)blockIdx.x * blockDim.x + threadIdx.x;
  long st = (long)gridDim.x * blockDim.x;
  for (; i < n; i += st) p[i] = 0.f;
}

__global__ void k_zero_out(float* outp, float* houtp) {
  int idx = blockIdx.x * blockDim.x + threadIdx.x;
  if (idx >= B_ * S_ * 67) return;
  int b = idx / (S_ * 67);
  int rem = idx % (S_ * 67);
  int s = rem / 67, c = rem % 67;
  if (c < 3) outp[((size_t)b * 123 * S_ + s) * 3 + c] = 0.f;
  else       houtp[((size_t)b * 123 * S_ + s) * H_ + (c - 3)] = 0.f;
}

/* ---------------- step phase 1: s_i, s_f gates -> h_g, c_g ---------------- */
__global__ void k_s1(const float* __restrict__ h_after, const float* __restrict__ h_before,
                     const _Float16* siWh, const _Float16* sfWh,
                     const float* si_b, const float* sf_b,
                     const float* h_seg, const float* c_seg,
                     float* h_g, float* c_g) {
  int wid = (blockIdx.x * blockDim.x + threadIdx.x) >> 5;
  int rt = wid >> 2, ct = wid & 3;
  int rbase = rt * 16, cbase = ct * 16;
  v8f ai = {}, af = {};
#pragma unroll
  for (int kc = 0; kc < 4; ++kc) {
    const float* Asrc = (kc < 2) ? h_after : h_before;
    int akb = (kc & 1) * 32;
    v16h a  = frag_a_f32(Asrc, rbase, H_, akb);
    v16h bi = frag_b_h(siWh, cbase, 2 * H_, kc * 32);
    v16h bf = frag_b_h(sfWh, cbase, 2 * H_, kc * 32);
    ai = wmma32(a, bi, ai);
    af = wmma32(a, bf, af);
  }
  int lane = threadIdx.x & 31;
  int n = lane & 15, mo = (lane < 16) ? 0 : 8;
#pragma unroll
  for (int r = 0; r < 8; ++r) {
    int row = rbase + mo + r;
    int col = cbase + n;
    float si = sigm(ai[r] + si_b[col]);
    float sf = sigm(af[r] + sf_b[col]);
    int ix = row * H_ + col;
    h_g[ix] = h_seg[ix] * si;
    c_g[ix] = c_seg[ix] * sf;
  }
}

/* ---------------- step phase 2: segment LSTM ---------------- */
__global__ void k_s2(const float* co, int t, const int* tab,
                     const _Float16* Wih, const _Float16* Whh,
                     const float* bih, const float* bhh,
                     const float* h_g, const float* c_g,
                     float* h_seg_n, float* c_seg_n) {
  int wid = (blockIdx.x * blockDim.x + threadIdx.x) >> 5;
  int rt = wid >> 2, hc = wid & 3;
  int rbase = rt * 16;
  v8f a0 = {}, a1 = {}, a2 = {}, a3 = {};
  {
    v16h ax = frag_a_xseg(co, t, rbase, tab);
    a0 = wmma32(ax, frag_b_h(Wih, 0 * H_ + hc * 16, 32, 0), a0);
    a1 = wmma32(ax, frag_b_h(Wih, 1 * H_ + hc * 16, 32, 0), a1);
    a2 = wmma32(ax, frag_b_h(Wih, 2 * H_ + hc * 16, 32, 0), a2);
    a3 = wmma32(ax, frag_b_h(Wih, 3 * H_ + hc * 16, 32, 0), a3);
  }
#pragma unroll
  for (int kb = 0; kb < H_; kb += 32) {
    v16h ah = frag_a_f32(h_g, rbase, H_, kb);
    a0 = wmma32(ah, frag_b_h(Whh, 0 * H_ + hc * 16, H_, kb), a0);
    a1 = wmma32(ah, frag_b_h(Whh, 1 * H_ + hc * 16, H_, kb), a1);
    a2 = wmma32(ah, frag_b_h(Whh, 2 * H_ + hc * 16, H_, kb), a2);
    a3 = wmma32(ah, frag_b_h(Whh, 3 * H_ + hc * 16, H_, kb), a3);
  }
  int lane = threadIdx.x & 31;
  int n = lane & 15, mo = (lane < 16) ? 0 : 8;
#pragma unroll
  for (int r = 0; r < 8; ++r) {
    int row = rbase + mo + r;
    int col = hc * 16 + n;
    float gi = sigm(a0[r] + bih[col] + bhh[col]);
    float gf = sigm(a1[r] + bih[H_ + col] + bhh[H_ + col]);
    float gg = tanhf(a2[r] + bih[2 * H_ + col] + bhh[2 * H_ + col]);
    float go = sigm(a3[r] + bih[3 * H_ + col] + bhh[3 * H_ + col]);
    int ix = row * H_ + col;
    float c2 = gf * c_g[ix] + gi * gg;
    h_seg_n[ix] = go * tanhf(c2);
    c_seg_n[ix] = c2;
  }
}

/* ---------------- step phase 3: inter LSTM ---------------- */
__global__ void k_s3(const float* co, int t, const int* tab,
                     const _Float16* Wih, const _Float16* Whh,
                     const float* bih, const float* bhh,
                     const float* h_int, const float* c_int,
                     float* h_i, float* c_i) {
  int wid = (blockIdx.x * blockDim.x + threadIdx.x) >> 5;
  int rt = wid / 28, hc = wid % 28;
  int rbase = rt * 16;
  v8f a0 = {}, a1 = {}, a2 = {}, a3 = {};
  {
    v16h ax = frag_a_xint(co, t, rbase, tab);
    a0 = wmma32(ax, frag_b_h(Wih, 0 * IH_ + hc * 16, 32, 0), a0);
    a1 = wmma32(ax, frag_b_h(Wih, 1 * IH_ + hc * 16, 32, 0), a1);
    a2 = wmma32(ax, frag_b_h(Wih, 2 * IH_ + hc * 16, 32, 0), a2);
    a3 = wmma32(ax, frag_b_h(Wih, 3 * IH_ + hc * 16, 32, 0), a3);
  }
  for (int kb = 0; kb < IH_; kb += 32) {
    v16h ah = frag_a_f32(h_int, rbase, IH_, kb);
    a0 = wmma32(ah, frag_b_h(Whh, 0 * IH_ + hc * 16, IH_, kb), a0);
    a1 = wmma32(ah, frag_b_h(Whh, 1 * IH_ + hc * 16, IH_, kb), a1);
    a2 = wmma32(ah, frag_b_h(Whh, 2 * IH_ + hc * 16, IH_, kb), a2);
    a3 = wmma32(ah, frag_b_h(Whh, 3 * IH_ + hc * 16, IH_, kb), a3);
  }
  int lane = threadIdx.x & 31;
  int n = lane & 15, mo = (lane < 16) ? 0 : 8;
#pragma unroll
  for (int r = 0; r < 8; ++r) {
    int row = rbase + mo + r;
    int col = hc * 16 + n;
    float gi = sigm(a0[r] + bih[col] + bhh[col]);
    float gf = sigm(a1[r] + bih[IH_ + col] + bhh[IH_ + col]);
    float gg = tanhf(a2[r] + bih[2 * IH_ + col] + bhh[2 * IH_ + col]);
    float go = sigm(a3[r] + bih[3 * IH_ + col] + bhh[3 * IH_ + col]);
    int ix = row * IH_ + col;
    float c2 = gf * c_int[ix] + gi * gg;
    h_i[ix] = go * tanhf(c2);
    c_i[ix] = c2;
  }
}

/* ---------------- step phase 4: scatter carries + h_outputs ---------------- */
__global__ void k_s4(int t, const int* tab,
                     const float* h_seg_n, const float* c_seg_n,
                     const float* h_i, const float* c_i,
                     float* h_int_n, float* c_int_n,
                     float* h_after, float* h_before,
                     float* houtp) {
  int idx = blockIdx.x * blockDim.x + threadIdx.x;
  if (idx >= B_ * S_ * H_) return;
  int h = idx & 63;
  int s = (idx >> 6) % S_;
  int b = idx / (S_ * H_);
  int si_ = tab[TAB_SEGIDX + s];
  int u   = tab[TAB_UNITOF + s];
  float vh, vc;
  if (si_ >= 0) { int ix = (b * SEGN + si_) * H_ + h; vh = h_seg_n[ix]; vc = c_seg_n[ix]; }
  else          { int ix = b * IH_ + u * H_ + h;      vh = h_i[ix];     vc = c_i[ix]; }
  if (t >= 5) houtp[(((size_t)b * 123 + (t - 4)) * S_ + s) * H_ + h] = vh;
  if (u >= 0) { int ix = b * IH_ + u * H_ + h; h_int_n[ix] = vh; c_int_n[ix] = vc; }
  int ka = tab[TAB_AFTDST + s];
  if (ka >= 0) h_after[(b * SEGN + ka) * H_ + h] = vh;
  int kb = tab[TAB_BEFDST + s];
  if (kb >= 0) h_before[(b * SEGN + kb) * H_ + h] = vh;
  if (s == 0) {
    h_after[(b * SEGN + (SEGN - 1)) * H_ + h] = 0.f;
    h_before[(b * SEGN + 0) * H_ + h] = 0.f;
    int kz = tab[TAB_BN + 1];
    if (kz >= 0 && kz < SEGN) h_before[(b * SEGN + kz) * H_ + h] = 0.f;
  }
}

/* ---------------- output MLP: relu(sel@W1^T+b1)@W2^T+b2 ---------------- */
__global__ void k_o1(const float* houtp, const _Float16* W1h,
                     const float* b1, const float* W2, const float* b2,
                     float* out_buf) {
  int wid = (blockIdx.x * blockDim.x + threadIdx.x) >> 5;
  int rbase = wid * 16;
  v8f acc[8];
#pragma unroll
  for (int ct = 0; ct < 8; ++ct)
#pragma unroll
    for (int r = 0; r < 8; ++r) acc[ct][r] = 0.f;
#pragma unroll
  for (int kb = 0; kb < H_; kb += 32) {
    v16h a = frag_a_sel(houtp, rbase, kb);
#pragma unroll
    for (int ct = 0; ct < 8; ++ct)
      acc[ct] = wmma32(a, frag_b_h(W1h, ct * 16, H_, kb), acc[ct]);
  }
  int lane = threadIdx.x & 31;
  int n = lane & 15, mo = (lane < 16) ? 0 : 8;
  float part[8];
#pragma unroll
  for (int r = 0; r < 8; ++r) part[r] = 0.f;
#pragma unroll
  for (int ct = 0; ct < 8; ++ct) {
    float b1v = b1[ct * 16 + n];
    float w2v = W2[ct * 16 + n];
#pragma unroll
    for (int r = 0; r < 8; ++r) {
      float hv = acc[ct][r] + b1v;
      part[r] += (hv > 0.f ? hv : 0.f) * w2v;
    }
  }
#pragma unroll
  for (int mk = 8; mk >= 1; mk >>= 1)
#pragma unroll
    for (int r = 0; r < 8; ++r)
      part[r] += __shfl_xor(part[r], mk, 32);
  if (n == 0) {
    float bb = b2[0];
#pragma unroll
    for (int r = 0; r < 8; ++r) out_buf[rbase + mo + r] = part[r] + bb;
  }
}

/* ---------------- final assembly: In / number_calc / outputs ---------------- */
__global__ void k_o2(const float* co, const float* out_buf, float* outp, const int* tab) {
  int idx = blockIdx.x * blockDim.x + threadIdx.x;
  if (idx >= N1) return;
  int k = idx / (B_ * S_);
  int rem = idx % (B_ * S_);
  int b = rem / S_, s = rem % S_;
  int bn1 = tab[TAB_BN + 1], bn3 = tab[TAB_BN + 3];
  int bn4 = tab[TAB_BN + 4], bn6 = tab[TAB_BN + 6];
  int rowbase = k * (B_ * S_) + b * S_;
  float o = out_buf[idx];
  float In = 0.f;
  int m = tab[TAB_INMAP + s];
  if (m >= 0) In += out_buf[rowbase + m];
  if (s == 0 || s == bn1 + 1) In += co[(((size_t)b * T_ + (k + 6)) * S_ + s) * 3 + 1];
  if (s == bn6) In += out_buf[rowbase + bn1] + out_buf[rowbase + bn3];
  if (s == bn4) In += out_buf[rowbase + bn6];
  float num = co[(((size_t)b * T_ + (k + 5)) * S_ + s) * 3 + 2] + In - o;
  size_t ob = ((size_t)(b * 123 + (k + 1)) * S_ + s) * 3;
  outp[ob] = o; outp[ob + 1] = In; outp[ob + 2] = num;
}

/* ================= host ================= */
extern "C" void kernel_launch(void* const* d_in, const int* in_sizes, int n_in,
                              void* d_out, int out_size, void* d_ws, size_t ws_size,
                              hipStream_t stream) {
  (void)in_sizes; (void)n_in; (void)out_size; (void)ws_size;
  const float*     co    = (const float*)d_in[0];
  const long long* bn64  = (const long long*)d_in[1];
  const float* segWih = (const float*)d_in[2];
  const float* segWhh = (const float*)d_in[3];
  const float* segbih = (const float*)d_in[4];
  const float* segbhh = (const float*)d_in[5];
  const float* sfW    = (const float*)d_in[6];
  const float* sfb    = (const float*)d_in[7];
  const float* siW    = (const float*)d_in[8];
  const float* sib    = (const float*)d_in[9];
  const float* intWih = (const float*)d_in[10];
  const float* intWhh = (const float*)d_in[11];
  const float* intbih = (const float*)d_in[12];
  const float* intbhh = (const float*)d_in[13];
  const float* outW1  = (const float*)d_in[14];
  const float* outb1  = (const float*)d_in[15];
  const float* outW2  = (const float*)d_in[16];
  const float* outb2  = (const float*)d_in[17];

  float* outp  = (float*)d_out;
  float* houtp = outp + (size_t)B_ * 123 * S_ * 3;

  char* base = (char*)d_ws;
  size_t off = 0;
  auto alloc = [&](size_t bytes) -> void* {
    void* p = base + off;
    off += (bytes + 255) & ~(size_t)255;
    return p;
  };

  int* tab = (int*)alloc(TAB_TOTAL * sizeof(int));
  _Float16* siWh    = (_Float16*)alloc((size_t)64 * 128 * 2);
  _Float16* sfWh    = (_Float16*)alloc((size_t)64 * 128 * 2);
  _Float16* segWihh = (_Float16*)alloc((size_t)256 * 32 * 2);
  _Float16* segWhhh = (_Float16*)alloc((size_t)256 * 64 * 2);
  _Float16* intWihh = (_Float16*)alloc((size_t)1792 * 32 * 2);
  _Float16* intWhhh = (_Float16*)alloc((size_t)1792 * 448 * 2);
  _Float16* outW1h  = (_Float16*)alloc((size_t)128 * 64 * 2);

  const size_t SEG = (size_t)NSEG * H_;  /* 393216, multiple of 64 -> 256B chunks */
  const size_t INT = (size_t)B_ * IH_;   /* 28672 */
  float* zbase    = (float*)alloc(0);
  float* h_seg0   = (float*)alloc(SEG * 4);
  float* c_seg0   = (float*)alloc(SEG * 4);
  float* h_after  = (float*)alloc(SEG * 4);
  float* h_before = (float*)alloc(SEG * 4);
  float* h_int0   = (float*)alloc(INT * 4);
  float* c_int0   = (float*)alloc(INT * 4);
  long   zcount   = (long)(4 * SEG + 2 * INT);
  float* h_seg1   = (float*)alloc(SEG * 4);
  float* c_seg1   = (float*)alloc(SEG * 4);
  float* h_g      = (float*)alloc(SEG * 4);
  float* c_g      = (float*)alloc(SEG * 4);
  float* h_int1   = (float*)alloc(INT * 4);
  float* c_int1   = (float*)alloc(INT * 4);
  float* h_i      = (float*)alloc(INT * 4);
  float* c_i      = (float*)alloc(INT * 4);
  float* out_buf  = (float*)alloc((size_t)N1 * 4);

  k_topology<<<1, 1, 0, stream>>>(bn64, tab);
  k_prepw<<<512, 256, 0, stream>>>(siW, sfW, segWih, segWhh, intWih, intWhh, outW1,
                                   siWh, sfWh, segWihh, segWhhh, intWihh, intWhhh, outW1h);
  k_zero<<<2048, 256, 0, stream>>>(zbase, zcount);
  k_zero_out<<<(B_ * S_ * 67 + 255) / 256, 256, 0, stream>>>(outp, houtp);

  float *hs_c = h_seg0, *cs_c = c_seg0, *hs_n = h_seg1, *cs_n = c_seg1;
  float *hi_c = h_int0, *ci_c = c_int0, *hi_n = h_int1, *ci_n = c_int1;
  for (int t = 0; t < TEMPORAL; ++t) {
    k_s1<<<192, 256, 0, stream>>>(h_after, h_before, siWh, sfWh, sib, sfb,
                                  hs_c, cs_c, h_g, c_g);
    k_s2<<<192, 256, 0, stream>>>(co, t, tab, segWihh, segWhhh, segbih, segbhh,
                                  h_g, c_g, hs_n, cs_n);
    k_s3<<<28, 128, 0, stream>>>(co, t, tab, intWihh, intWhhh, intbih, intbhh,
                                 hi_c, ci_c, h_i, c_i);
    k_s4<<<1600, 256, 0, stream>>>(t, tab, hs_n, cs_n, h_i, c_i,
                                   hi_n, ci_n, h_after, h_before, houtp);
    float* tmp;
    tmp = hs_c; hs_c = hs_n; hs_n = tmp;
    tmp = cs_c; cs_c = cs_n; cs_n = tmp;
    tmp = hi_c; hi_c = hi_n; hi_n = tmp;
    tmp = ci_c; ci_c = ci_n; ci_n = tmp;
  }
  k_o1<<<6100, 256, 0, stream>>>(houtp, outW1h, outb1, outW2, outb2, out_buf);
  k_o2<<<3050, 256, 0, stream>>>(co, out_buf, outp, tab);
}
